// GatedArch_37168646980319
// MI455X (gfx1250) — compile-verified, matrix operand order checked by reading, not run on hardware
//
#include <hip/hip_runtime.h>
#include <cstdint>
#include <cstddef>

// ---------------------------------------------------------------------------
// MI455X (gfx1250) gated-RNN implementation.
//  - bf16 V_WMMA_F32_16X16X32_BF16 for all GEMMs (fp32 accumulate)
//  - persistent scan kernel: weights staged once in LDS (WGP has 320KB),
//    h double-buffered in global (L2-resident: 64x1024), one grid barrier/step
// ---------------------------------------------------------------------------

typedef __bf16 bf16;
typedef __attribute__((ext_vector_type(16))) __bf16 bf16x16;
typedef __attribute__((ext_vector_type(8)))  float  f32x8;
typedef __attribute__((ext_vector_type(4)))  float  f32x4;

union Frag {
  bf16x16 v;
  f32x4   q[2];   // q[0] = elements 0..7, q[1] = elements 8..15
};

#define BBAT 64
#define TLEN 512
#define DDIM 1024
#define RDIM 1024
#define ODIM 1024
#define SCAN_BLOCKS 64

// A/B 16-bit fragment from a row-major matrix, computing against B^T:
// lane l -> row (l&15), K-half (l>>4); elements 0..7 = K{kh*8..kh*8+7},
// elements 8..15 = K{16+kh*8 ..}. Two 16B loads.
__device__ __forceinline__ bf16x16 load_frag_g(const bf16* p) {
  Frag f;
  f.q[0] = *(const f32x4*)(p);        // K offset +0
  f.q[1] = *(const f32x4*)(p + 16);   // K offset +16 elements (+32B)
  return f.v;
}

// LDS fragments are pre-swizzled compact: chunk0 at p, chunk1 at p+8 elements.
__device__ __forceinline__ bf16x16 load_frag_lds(const bf16* p) {
  Frag f;
  f.q[0] = *(const f32x4*)(p);
  f.q[1] = *(const f32x4*)(p + 8);
  return f.v;
}

// ---------------------------------------------------------------------------
__global__ void __launch_bounds__(256)
convert_f32_bf16_kernel(const float* __restrict__ in, bf16* __restrict__ out, size_t n) {
  size_t i = (size_t)blockIdx.x * blockDim.x + threadIdx.x;
  size_t stride = (size_t)gridDim.x * blockDim.x;
  for (; i < n; i += stride) out[i] = (bf16)in[i];
}

// out[j][k] = in[k][j] for square n x n (gives Whz^T so Mz = Wxz @ (Whz^T)^T).
__global__ void __launch_bounds__(256)
transpose_convert_kernel(const float* __restrict__ in, bf16* __restrict__ out, int n) {
  int idx = blockIdx.x * 256 + threadIdx.x;
  if (idx < n * n) {
    int j = idx / n;
    int k = idx - j * n;
    out[idx] = (bf16)in[k * n + j];
  }
}

// ---------------------------------------------------------------------------
// Generic C[M][N] = sum_k A[m][k] * B[n][k]  (+ bias[n]), A/B bf16 row-major
// with leading dim K, C fp32 row-major with leading dim N.
// Block = 8 waves; block tile 128(M) x 64(N); wave tile 16 x 64 (A reuse x4).
// ---------------------------------------------------------------------------
__global__ void __launch_bounds__(256)
gemm_bt_wmma_kernel(const bf16* __restrict__ A, const bf16* __restrict__ Bw,
                    const float* __restrict__ bias, float* __restrict__ C,
                    int M, int N, int K) {
  const int wave = threadIdx.x >> 5;
  const int lane = threadIdx.x & 31;
  const int m0 = blockIdx.y * 128 + wave * 16;
  if (m0 >= M) return;                       // wave-uniform
  const int n0 = blockIdx.x * 64;
  const int kh = lane >> 4;
  const int nl = lane & 15;

  const bf16* arow = A  + (size_t)(m0 + nl) * K + kh * 8;
  const bf16* br0  = Bw + (size_t)(n0 +  0 + nl) * K + kh * 8;
  const bf16* br1  = Bw + (size_t)(n0 + 16 + nl) * K + kh * 8;
  const bf16* br2  = Bw + (size_t)(n0 + 32 + nl) * K + kh * 8;
  const bf16* br3  = Bw + (size_t)(n0 + 48 + nl) * K + kh * 8;

  f32x8 acc0 = {}, acc1 = {}, acc2 = {}, acc3 = {};
  for (int k0 = 0; k0 < K; k0 += 32) {
    bf16x16 a  = load_frag_g(arow + k0);
    bf16x16 b0 = load_frag_g(br0 + k0);
    bf16x16 b1 = load_frag_g(br1 + k0);
    bf16x16 b2 = load_frag_g(br2 + k0);
    bf16x16 b3 = load_frag_g(br3 + k0);
    acc0 = __builtin_amdgcn_wmma_f32_16x16x32_bf16(false, a, false, b0, (short)0, acc0, false, false);
    acc1 = __builtin_amdgcn_wmma_f32_16x16x32_bf16(false, a, false, b1, (short)0, acc1, false, false);
    acc2 = __builtin_amdgcn_wmma_f32_16x16x32_bf16(false, a, false, b2, (short)0, acc2, false, false);
    acc3 = __builtin_amdgcn_wmma_f32_16x16x32_bf16(false, a, false, b3, (short)0, acc3, false, false);
  }

  float bv0 = 0.f, bv1 = 0.f, bv2 = 0.f, bv3 = 0.f;
  if (bias) {
    bv0 = bias[n0 + nl];      bv1 = bias[n0 + 16 + nl];
    bv2 = bias[n0 + 32 + nl]; bv3 = bias[n0 + 48 + nl];
  }
  #pragma unroll
  for (int e = 0; e < 8; ++e) {
    const int m = m0 + e + 8 * kh;           // C/D layout: VGPR e, lane half
    float* crow = C + (size_t)m * N + n0 + nl;
    crow[0]  = acc0[e] + bv0;
    crow[16] = acc1[e] + bv1;
    crow[32] = acc2[e] + bv2;
    crow[48] = acc3[e] + bv3;
  }
}

// ---------------------------------------------------------------------------
// Device-wide two-phase barrier (all SCAN_BLOCKS blocks are co-resident).
// ---------------------------------------------------------------------------
__device__ __forceinline__ void grid_sync(unsigned* bar, unsigned nblocks) {
  __threadfence();               // flush this thread's h stores device-wide
  __syncthreads();
  if (threadIdx.x == 0) {
    unsigned g = __hip_atomic_load(&bar[1], __ATOMIC_RELAXED, __HIP_MEMORY_SCOPE_AGENT);
    unsigned arrived =
        __hip_atomic_fetch_add(&bar[0], 1u, __ATOMIC_ACQ_REL, __HIP_MEMORY_SCOPE_AGENT) + 1u;
    if (arrived == nblocks) {
      __hip_atomic_store(&bar[0], 0u, __ATOMIC_RELAXED, __HIP_MEMORY_SCOPE_AGENT);
      __hip_atomic_store(&bar[1], g + 1u, __ATOMIC_RELEASE, __HIP_MEMORY_SCOPE_AGENT);
      (void)__hip_atomic_load(&bar[1], __ATOMIC_ACQUIRE, __HIP_MEMORY_SCOPE_AGENT);
    } else {
      while (__hip_atomic_load(&bar[1], __ATOMIC_ACQUIRE, __HIP_MEMORY_SCOPE_AGENT) == g) {
        __builtin_amdgcn_s_sleep(8);
      }
    }
  }
  __syncthreads();
}

// ---------------------------------------------------------------------------
// Persistent scan. Each of 64 workgroups owns 16 columns of R=1024.
// LDS: Whh slice (32KB, fragment order) + Mz slice (32KB) + z exchange (4KB).
// Waves 0-3: hb tiles (relu branch);  waves 4-7: z tiles (sigmoid branch).
// h double-buffered (fp32 for the gate math + bf16 mirror as next A operand).
// ---------------------------------------------------------------------------
__global__ void __launch_bounds__(256)
gated_scan_kernel(const bf16* __restrict__ WhhB, const bf16* __restrict__ MzB,
                  const float* __restrict__ xh, const float* __restrict__ xz,
                  float* __restrict__ h32a, float* __restrict__ h32b,
                  bf16* __restrict__ hbfa, bf16* __restrict__ hbfb,
                  unsigned* __restrict__ bar) {
  extern __shared__ char smem[];
  bf16*  sWhh = (bf16*)smem;                  // 32 KB
  bf16*  sMz  = (bf16*)(smem + 32 * 1024);    // 32 KB
  float* sZ   = (float*)(smem + 64 * 1024);   //  4 KB

  const int tid = threadIdx.x;
  const int wg  = blockIdx.x;                 // owns columns [wg*16, wg*16+16)

  // Stage weight slices once, pre-swizzled into WMMA B-fragment order:
  // per k-step kk, lane l holds row n=wg*16+(l&15), K-half (l>>4).
  for (int idx = tid; idx < 32 * 32; idx += 256) {
    const int kk = idx >> 5;
    const int ln = idx & 31;
    const int n  = wg * 16 + (ln & 15);
    const int kh = ln >> 4;
    const size_t goff = (size_t)n * RDIM + kk * 32 + kh * 8;
    f32x4* dw = (f32x4*)(sWhh + (size_t)kk * 512 + ln * 16);
    dw[0] = *(const f32x4*)(WhhB + goff);
    dw[1] = *(const f32x4*)(WhhB + goff + 16);
    f32x4* dm = (f32x4*)(sMz + (size_t)kk * 512 + ln * 16);
    dm[0] = *(const f32x4*)(MzB + goff);
    dm[1] = *(const f32x4*)(MzB + goff + 16);
  }

  // h0 = 0 (buffer A is read at t=0).
  for (int i = wg * 256 + tid; i < BBAT * RDIM; i += SCAN_BLOCKS * 256) {
    h32a[i] = 0.0f;
    hbfa[i] = (bf16)0.0f;
  }
  grid_sync(bar, gridDim.x);

  const int wave = tid >> 5;
  const int lane = tid & 31;
  const int mt   = wave & 3;                  // batch tile (16 rows)
  const bool isZ = wave >= 4;                 // wave-uniform
  const bf16* sW = isZ ? sMz : sWhh;
  const int nl   = lane & 15;
  const int kh   = lane >> 4;
  const int arow = mt * 16 + nl;              // batch row for A fragment
  const int ncol = wg * 16 + nl;              // global column of this lane

  for (int t = 0; t < TLEN; ++t) {
    const bf16*  hin   = (t & 1) ? hbfb : hbfa;
    const float* hprev = (t & 1) ? h32b : h32a;
    float*       hout  = (t & 1) ? h32a : h32b;
    bf16*        houtb = (t & 1) ? hbfa : hbfb;

    const bf16* ap = hin + (size_t)arow * RDIM + kh * 8;
    f32x8 acc = {};
    #pragma unroll 4
    for (int kk = 0; kk < 32; ++kk) {
      bf16x16 a = load_frag_g(ap + kk * 32);                 // h (L2-resident)
      bf16x16 b = load_frag_lds(sW + kk * 512 + lane * 16);  // weight slice
      acc = __builtin_amdgcn_wmma_f32_16x16x32_bf16(false, a, false, b, (short)0,
                                                    acc, false, false);
    }

    if (isZ) {
      #pragma unroll
      for (int e = 0; e < 8; ++e) {
        const int ml = e + 8 * kh;
        const int b  = mt * 16 + ml;
        const float pre = acc[e] + xz[((size_t)b * TLEN + t) * RDIM + ncol];
        const float zv  = 1.0f / (1.0f + __expf(-pre));
        sZ[mt * 256 + ml * 16 + nl] = zv;
      }
    }
    __syncthreads();
    if (!isZ) {
      #pragma unroll
      for (int e = 0; e < 8; ++e) {
        const int ml = e + 8 * kh;
        const int b  = mt * 16 + ml;
        const size_t gi = (size_t)b * RDIM + ncol;
        float hb = acc[e] + xh[((size_t)b * TLEN + t) * RDIM + ncol];
        hb = fmaxf(hb, 0.0f);
        const float zv = sZ[mt * 256 + ml * 16 + nl];
        const float hn = zv * hprev[gi] + (1.0f - zv) * hb;
        hout[gi]  = hn;
        houtb[gi] = (bf16)hn;
      }
    }
    grid_sync(bar, gridDim.x);   // publish h(t) before anyone reads at t+1
  }
}

// ---------------------------------------------------------------------------
extern "C" void kernel_launch(void* const* d_in, const int* in_sizes, int n_in,
                              void* d_out, int out_size, void* d_ws, size_t ws_size,
                              hipStream_t stream) {
  (void)in_sizes; (void)n_in; (void)out_size; (void)ws_size;
  const float* x   = (const float*)d_in[0];
  const float* Wxh = (const float*)d_in[1];
  const float* bh  = (const float*)d_in[2];
  const float* Whh = (const float*)d_in[3];
  const float* Wxz = (const float*)d_in[4];
  const float* bz  = (const float*)d_in[5];
  const float* Whz = (const float*)d_in[6];
  const float* Wy  = (const float*)d_in[7];
  const float* by  = (const float*)d_in[8];
  float* out = (float*)d_out;

  char* ws = (char*)d_ws;
  size_t off = 0;
  auto take = [&](size_t bytes) -> char* {
    char* r = ws + off;
    off = (off + bytes + 255) & ~(size_t)255;
    return r;
  };

  bf16*  x_bf    = (bf16*)take((size_t)BBAT * TLEN * DDIM * sizeof(bf16));
  bf16*  Wxh_bf  = (bf16*)take((size_t)RDIM * DDIM * sizeof(bf16));
  bf16*  Wxz_bf  = (bf16*)take((size_t)RDIM * DDIM * sizeof(bf16));
  bf16*  Whh_bf  = (bf16*)take((size_t)RDIM * RDIM * sizeof(bf16));
  bf16*  WhzT_bf = (bf16*)take((size_t)RDIM * RDIM * sizeof(bf16));
  bf16*  Wy_bf   = (bf16*)take((size_t)ODIM * RDIM * sizeof(bf16));
  float* Mz_f32  = (float*)take((size_t)RDIM * RDIM * sizeof(float));
  bf16*  Mz_bf   = (bf16*)take((size_t)RDIM * RDIM * sizeof(bf16));
  float* xh      = (float*)take((size_t)BBAT * TLEN * RDIM * sizeof(float));
  float* xz      = (float*)take((size_t)BBAT * TLEN * RDIM * sizeof(float));
  float* h32a    = (float*)take((size_t)BBAT * RDIM * sizeof(float));
  float* h32b    = (float*)take((size_t)BBAT * RDIM * sizeof(float));
  bf16*  hbfa    = (bf16*)take((size_t)BBAT * RDIM * sizeof(bf16));
  bf16*  hbfb    = (bf16*)take((size_t)BBAT * RDIM * sizeof(bf16));
  unsigned* bar  = (unsigned*)take(256);

  // --- precision conversions -------------------------------------------------
  convert_f32_bf16_kernel<<<4096, 256, 0, stream>>>(x, x_bf, (size_t)BBAT * TLEN * DDIM);
  convert_f32_bf16_kernel<<<1024, 256, 0, stream>>>(Wxh, Wxh_bf, (size_t)RDIM * DDIM);
  convert_f32_bf16_kernel<<<1024, 256, 0, stream>>>(Wxz, Wxz_bf, (size_t)RDIM * DDIM);
  convert_f32_bf16_kernel<<<1024, 256, 0, stream>>>(Whh, Whh_bf, (size_t)RDIM * RDIM);
  convert_f32_bf16_kernel<<<1024, 256, 0, stream>>>(Wy, Wy_bf, (size_t)ODIM * RDIM);
  transpose_convert_kernel<<<(RDIM * RDIM + 255) / 256, 256, 0, stream>>>(Whz, WhzT_bf, RDIM);

  // --- Mz = Wxz @ Whz  (as Wxz · (Whz^T)^T), then to bf16 --------------------
  gemm_bt_wmma_kernel<<<dim3(RDIM / 64, (RDIM + 127) / 128), 256, 0, stream>>>(
      Wxz_bf, WhzT_bf, nullptr, Mz_f32, RDIM, RDIM, RDIM);
  convert_f32_bf16_kernel<<<1024, 256, 0, stream>>>(Mz_f32, Mz_bf, (size_t)RDIM * RDIM);

  // --- input-side GEMMs: xh = x@Wxh^T + bh ; xz = x@Wxz^T + bz ---------------
  gemm_bt_wmma_kernel<<<dim3(RDIM / 64, (BBAT * TLEN + 127) / 128), 256, 0, stream>>>(
      x_bf, Wxh_bf, bh, xh, BBAT * TLEN, RDIM, DDIM);
  gemm_bt_wmma_kernel<<<dim3(RDIM / 64, (BBAT * TLEN + 127) / 128), 256, 0, stream>>>(
      x_bf, Wxz_bf, bz, xz, BBAT * TLEN, RDIM, DDIM);

  // --- persistent recurrent scan --------------------------------------------
  hipMemsetAsync(bar, 0, 256, stream);
  const int scan_lds = 32 * 1024 + 32 * 1024 + 4 * 1024;   // 68 KB of 320 KB WGP LDS
  hipFuncSetAttribute(reinterpret_cast<const void*>(gated_scan_kernel),
                      hipFuncAttributeMaxDynamicSharedMemorySize, scan_lds);
  gated_scan_kernel<<<SCAN_BLOCKS, 256, scan_lds, stream>>>(
      Whh_bf, Mz_bf, xh, xz, h32a, h32b, hbfa, hbfb, bar);

  // --- out = h_final @ Wy^T + by  (T even -> final h lives in buffer A) ------
  gemm_bt_wmma_kernel<<<dim3(ODIM / 64, (BBAT + 127) / 128), 256, 0, stream>>>(
      hbfa, Wy_bf, by, out, BBAT, ODIM, RDIM);
}
